// CAPMemory_33148557591295
// MI455X (gfx1250) — matrix operand order repaired and checked by reading, compiled, not run on hardware
//
#include <hip/hip_runtime.h>
#include <hip/hip_bf16.h>
#include <math.h>

// ---------------- problem constants (match reference) ----------------
#define BSZ   1024          // batch
#define DDIM  2048          // feature dim
#define NCAM  8
#define LCLS  2048
#define NCOL  (NCAM * LCLS) // 16384 flattened centers
#define INV_T 20.0f         // 1/0.05
#define KNEG  50

// ---------------- GEMM tiling ----------------
#define TM 128
#define TN 128
#define TK 32
#define KPAD 40             // 32 + 8 pad -> 80B row stride, conflict-free b128

typedef __attribute__((ext_vector_type(16))) __bf16 v16bf;
typedef __attribute__((ext_vector_type(8)))  __bf16 v8bf;
typedef __attribute__((ext_vector_type(8)))  float  v8f;

// =====================================================================
// Kernel 0: zero the 24-float stats block (ce_sum[8], cnt[8], inter[8])
// =====================================================================
__global__ void cap_init(float* __restrict__ stats) {
    if (threadIdx.x < 24) stats[threadIdx.x] = 0.0f;
}

// =====================================================================
// Kernel 1: rnorm[b] = 1/||inputs[b]||
// =====================================================================
__global__ __launch_bounds__(256) void cap_rnorm(const float* __restrict__ X,
                                                 float* __restrict__ rnorm) {
    const int b = blockIdx.x, t = threadIdx.x;
    const float* xp = X + (size_t)b * DDIM;
    float s = 0.0f;
    for (int i = t; i < DDIM; i += 256) { float v = xp[i]; s += v * v; }
    __shared__ float red[256];
    red[t] = s; __syncthreads();
    for (int off = 128; off > 0; off >>= 1) {
        if (t < off) red[t] += red[t + off];
        __syncthreads();
    }
    if (t == 0) rnorm[b] = 1.0f / sqrtf(red[0]);
}

// =====================================================================
// Kernel 2: all_log[b, j] = x_hat[b,:] . Mem[j,:]   (j = c*L + l)
// fp32-accurate GEMM via bf16x3 split on v_wmma_f32_16x16x32_bf16.
// =====================================================================
__global__ __launch_bounds__(256) void cap_gemm(const float* __restrict__ X,
                                                const float* __restrict__ rnorm,
                                                const float* __restrict__ Mem,
                                                float* __restrict__ Out) {
    __shared__ __align__(16) __bf16 sAhi[TM][KPAD];
    __shared__ __align__(16) __bf16 sAlo[TM][KPAD];
    __shared__ __align__(16) __bf16 sBhi[TN][KPAD];
    __shared__ __align__(16) __bf16 sBlo[TN][KPAD];

    const int tid  = threadIdx.x;
    const int lane = tid & 31;
    const int wave = tid >> 5;        // 0..7
    const int wm   = wave >> 1;       // 0..3 : 32-row patch
    const int wn   = wave & 1;        // 0..1 : 64-col patch
    const int half = lane >> 4;       // 0/1
    const int l16  = lane & 15;
    const int row0 = blockIdx.y * TM;
    const int col0 = blockIdx.x * TN;

    // global staging assignment: 2 threads per tile-row, 16 floats each
    const int lr = tid >> 1;           // 0..127
    const int lc = (tid & 1) * 16;     // 0 or 16

    const float rn = rnorm[row0 + lr];
    const float* ap = X   + (size_t)(row0 + lr) * DDIM + lc;
    const float* bp = Mem + (size_t)(col0 + lr) * DDIM + lc;

    v8f acc[2][4];
    const v8f vzero = {0.f,0.f,0.f,0.f,0.f,0.f,0.f,0.f};
#pragma unroll
    for (int i = 0; i < 2; ++i)
#pragma unroll
        for (int j = 0; j < 4; ++j) acc[i][j] = vzero;

    for (int k0 = 0; k0 < DDIM; k0 += TK) {
        // ---- stage A/B tiles: load f32, normalize A, split hi/lo bf16 ----
        float4 a4[4], b4[4];
#pragma unroll
        for (int q = 0; q < 4; ++q) {
            a4[q] = *(const float4*)(ap + k0 + q * 4);
            b4[q] = *(const float4*)(bp + k0 + q * 4);
        }
        v8bf ah[2], al[2], bh[2], bl[2];
#pragma unroll
        for (int q = 0; q < 4; ++q) {
            const float* af = (const float*)&a4[q];
            const float* bf = (const float*)&b4[q];
#pragma unroll
            for (int e = 0; e < 4; ++e) {
                const int ii = q * 4 + e;        // 0..15
                const int vv = ii >> 3, ee = ii & 7;
                float va = af[e] * rn;
                __bf16 hA = (__bf16)va;
                ah[vv][ee] = hA;
                al[vv][ee] = (__bf16)(va - (float)hA);
                float vb = bf[e];
                __bf16 hB = (__bf16)vb;
                bh[vv][ee] = hB;
                bl[vv][ee] = (__bf16)(vb - (float)hB);
            }
        }
        *(v8bf*)&sAhi[lr][lc]     = ah[0];
        *(v8bf*)&sAhi[lr][lc + 8] = ah[1];
        *(v8bf*)&sAlo[lr][lc]     = al[0];
        *(v8bf*)&sAlo[lr][lc + 8] = al[1];
        *(v8bf*)&sBhi[lr][lc]     = bh[0];
        *(v8bf*)&sBhi[lr][lc + 8] = bh[1];
        *(v8bf*)&sBlo[lr][lc]     = bl[0];
        *(v8bf*)&sBlo[lr][lc + 8] = bl[1];
        __syncthreads();

        // prefetch next K tile while math runs (global_prefetch_b8)
        if (k0 + TK < DDIM) {
            __builtin_prefetch(ap + k0 + TK, 0, 0);
            __builtin_prefetch(bp + k0 + TK, 0, 0);
        }

        // ---- fragment loads (ISA 7.12.2 16-bit A 16x32 / B 32x16 layouts) ----
        v16bf afh[2], afl[2], bfh[4], bfl[4];
        const int akb0 = half * 8;        // lanes 0-15: K 0..7 ; 16-31: K 8..15
        const int akb1 = 16 + half * 8;   // lanes 0-15: K16..23; 16-31: K24..31
#pragma unroll
        for (int tm = 0; tm < 2; ++tm) {
            const int m = wm * 32 + tm * 16 + l16;
            v8bf h0 = *(const v8bf*)&sAhi[m][akb0];
            v8bf h1 = *(const v8bf*)&sAhi[m][akb1];
            afh[tm] = __builtin_shufflevector(h0, h1, 0,1,2,3,4,5,6,7,8,9,10,11,12,13,14,15);
            v8bf l0 = *(const v8bf*)&sAlo[m][akb0];
            v8bf l1 = *(const v8bf*)&sAlo[m][akb1];
            afl[tm] = __builtin_shufflevector(l0, l1, 0,1,2,3,4,5,6,7,8,9,10,11,12,13,14,15);
        }
        const int bkb = half * 8;         // lanes 0-15: K 0..7,16..23 ; 16-31: K 8..15,24..31
#pragma unroll
        for (int tn = 0; tn < 4; ++tn) {
            const int n = wn * 64 + tn * 16 + l16;
            v8bf h0 = *(const v8bf*)&sBhi[n][bkb];
            v8bf h1 = *(const v8bf*)&sBhi[n][bkb + 16];
            bfh[tn] = __builtin_shufflevector(h0, h1, 0,1,2,3,4,5,6,7,8,9,10,11,12,13,14,15);
            v8bf l0 = *(const v8bf*)&sBlo[n][bkb];
            v8bf l1 = *(const v8bf*)&sBlo[n][bkb + 16];
            bfl[tn] = __builtin_shufflevector(l0, l1, 0,1,2,3,4,5,6,7,8,9,10,11,12,13,14,15);
        }

        // ---- bf16x3: acc += lo*hi + hi*lo + hi*hi  (f32 accumulate) ----
#pragma unroll
        for (int tm = 0; tm < 2; ++tm)
#pragma unroll
            for (int tn = 0; tn < 4; ++tn) {
                v8f a = acc[tm][tn];
                a = __builtin_amdgcn_wmma_f32_16x16x32_bf16(false, afl[tm], false, bfh[tn], (short)0, a, false, false);
                a = __builtin_amdgcn_wmma_f32_16x16x32_bf16(false, afh[tm], false, bfl[tn], (short)0, a, false, false);
                a = __builtin_amdgcn_wmma_f32_16x16x32_bf16(false, afh[tm], false, bfh[tn], (short)0, a, false, false);
                acc[tm][tn] = a;
            }
        __syncthreads();
    }

    // ---- store C (VGPR i: lanes 0-15 -> M=i, lanes 16-31 -> M=i+8) ----
#pragma unroll
    for (int tm = 0; tm < 2; ++tm)
#pragma unroll
        for (int tn = 0; tn < 4; ++tn) {
            const int r = row0 + wm * 32 + tm * 16 + half * 8;
            const int c = col0 + wn * 64 + tn * 16 + l16;
            float* op = Out + (size_t)r * NCOL + c;
#pragma unroll
            for (int i = 0; i < 8; ++i) op[(size_t)i * NCOL] = acc[tm][tn][i];
        }
}

// =====================================================================
// Kernel 3: intra-camera CE against own-camera memory (per sample)
// =====================================================================
__global__ __launch_bounds__(256) void cap_intra(const float* __restrict__ allog,
                                                 const int* __restrict__ indexes,
                                                 const int* __restrict__ cams_all,
                                                 const int* __restrict__ labels_all,
                                                 float* __restrict__ ce_sum,
                                                 float* __restrict__ cnt) {
    const int b = blockIdx.x, t = threadIdx.x;
    const int idx = indexes[b];
    const int cam = cams_all[idx];
    const int tgt = labels_all[idx];
    const float* row = allog + (size_t)b * NCOL + (size_t)cam * LCLS;
    __shared__ float red[256];

    float m = -3.4e38f;
    for (int i = t; i < LCLS; i += 256) m = fmaxf(m, row[i]);
    red[t] = m; __syncthreads();
    for (int off = 128; off > 0; off >>= 1) {
        if (t < off) red[t] = fmaxf(red[t], red[t + off]);
        __syncthreads();
    }
    m = red[0]; __syncthreads();

    float s = 0.0f;
    for (int i = t; i < LCLS; i += 256) s += expf((row[i] - m) * INV_T);
    red[t] = s; __syncthreads();
    for (int off = 128; off > 0; off >>= 1) {
        if (t < off) red[t] += red[t + off];
        __syncthreads();
    }
    if (t == 0) {
        const float lse = m * INV_T + logf(red[0]);
        const float ce  = lse - row[tgt] * INV_T;
        atomicAdd(&ce_sum[cam], ce);
        atomicAdd(&cnt[cam], 1.0f);
    }
}

// =====================================================================
// Kernel 4: inter-camera loss — 8 positives + top-50 hard negatives.
// Row cached in 64KB dynamic LDS; 50th-largest found by 32-step bisection.
// =====================================================================
__global__ __launch_bounds__(256) void cap_inter(const float* __restrict__ allog,
                                                 const int* __restrict__ indexes,
                                                 const int* __restrict__ cams_all,
                                                 const int* __restrict__ labels_all,
                                                 float* __restrict__ inter_sum) {
    extern __shared__ float sv[];        // NCOL floats
    __shared__ float red[256];
    __shared__ int   ired[256];
    __shared__ float spos[NCAM];

    const int b = blockIdx.x, t = threadIdx.x;
    const int idx = indexes[b];
    const int cam = cams_all[idx];
    const int tgt = labels_all[idx];
    const float* row = allog + (size_t)b * NCOL;

    for (int i = t; i < NCOL; i += 256) sv[i] = row[i];
    __syncthreads();
    if (t < NCAM) { spos[t] = sv[t * LCLS + tgt]; sv[t * LCLS + tgt] = -10000.0f; }
    __syncthreads();

    // global max over positives + masked negatives (softmax stabilizer)
    float m = -3.4e38f;
    for (int i = t; i < NCOL; i += 256) m = fmaxf(m, sv[i]);
    if (t < NCAM) m = fmaxf(m, spos[t]);
    red[t] = m; __syncthreads();
    for (int off = 128; off > 0; off >>= 1) {
        if (t < off) red[t] = fmaxf(red[t], red[t + off]);
        __syncthreads();
    }
    const float gmax = red[0];
    __syncthreads();

    // bisect the top-KNEG threshold (sims of unit vectors lie in [-1,1])
    float lo = -2.0f, hi = 2.0f;
    for (int it = 0; it < 32; ++it) {
        const float mid = 0.5f * (lo + hi);
        int c = 0;
        for (int i = t; i < NCOL; i += 256) c += (sv[i] > mid) ? 1 : 0;
        ired[t] = c; __syncthreads();
        for (int off = 128; off > 0; off >>= 1) {
            if (t < off) ired[t] += ired[t + off];
            __syncthreads();
        }
        const int total = ired[0];
        __syncthreads();
        if (total >= KNEG) lo = mid; else hi = mid;
    }

    // sum exp over the strict top region; fill remainder at the threshold
    int   c = 0;
    float s = 0.0f;
    for (int i = t; i < NCOL; i += 256) {
        const float v = sv[i];
        if (v > hi) { ++c; s += expf((v - gmax) * INV_T); }
    }
    ired[t] = c; red[t] = s; __syncthreads();
    for (int off = 128; off > 0; off >>= 1) {
        if (t < off) { ired[t] += ired[t + off]; red[t] += red[t + off]; }
        __syncthreads();
    }
    if (t == 0) {
        float negExp = red[0];
        float rem = (float)(KNEG - ired[0]);
        if (rem < 0.0f) rem = 0.0f;
        negExp += rem * expf((lo - gmax) * INV_T);
        float pe = 0.0f, pm = 0.0f;
#pragma unroll
        for (int cc = 0; cc < NCAM; ++cc) {
            pe += expf((spos[cc] - gmax) * INV_T);
            pm += spos[cc];
        }
        const float lse = gmax * INV_T + logf(pe + negExp);
        const float per = lse - (pm * (1.0f / NCAM)) * INV_T;
        atomicAdd(&inter_sum[cam], per);
    }
}

// =====================================================================
// Kernel 5: finalize the two scalar losses
// =====================================================================
__global__ void cap_final(const float* __restrict__ stats, float* __restrict__ out) {
    if (threadIdx.x == 0) {
        const float* ce_sum = stats;
        const float* cnt    = stats + 8;
        const float* inter  = stats + 16;
        float li = 0.0f, le = 0.0f;
        for (int c = 0; c < NCAM; ++c) {
            const float cc = fmaxf(cnt[c], 1.0f);
            li += ce_sum[c] / cc;
            le += inter[c] / cc;
        }
        out[0] = li;
        out[1] = le * 0.5f;   // LOSS_WEIGHT
    }
}

// =====================================================================
extern "C" void kernel_launch(void* const* d_in, const int* in_sizes, int n_in,
                              void* d_out, int out_size, void* d_ws, size_t ws_size,
                              hipStream_t stream) {
    const float* X      = (const float*)d_in[0];   // [B, D]
    const float* Mem    = (const float*)d_in[1];   // [C, L, D] == [NCOL, D]
    const int* indexes  = (const int*)d_in[2];     // [B]
    const int* cams     = (const int*)d_in[3];     // [N]
    const int* labels   = (const int*)d_in[4];     // [N]
    float* out = (float*)d_out;                    // [2]

    float* rnorm = (float*)d_ws;                       // 1024
    float* allog = rnorm + BSZ;                        // 1024*16384
    float* stats = allog + (size_t)BSZ * NCOL;         // 24: ce_sum|cnt|inter

    cap_init<<<1, 32, 0, stream>>>(stats);
    cap_rnorm<<<BSZ, 256, 0, stream>>>(X, rnorm);

    dim3 gg(NCOL / TN, BSZ / TM);                      // 128 x 8 = 1024 blocks
    cap_gemm<<<gg, 256, 0, stream>>>(X, rnorm, Mem, allog);

    cap_intra<<<BSZ, 256, 0, stream>>>(allog, indexes, cams, labels, stats, stats + 8);
    cap_inter<<<BSZ, 256, NCOL * sizeof(float), stream>>>(allog, indexes, cams, labels, stats + 16);
    cap_final<<<1, 32, 0, stream>>>(stats, out);
}
